// MultiHeadSelfAttention_36472862277880
// MI455X (gfx1250) — matrix operand order repaired and checked by reading, not compiled, not run on hardware
//
#include <hip/hip_runtime.h>

// ---------- constants ----------
#define BATCH 4
#define SEQ   2048
#define DM    1024
#define NH    16
#define DH    64
#define MTOT  (BATCH * SEQ)      // 8192
#define N_QKV (3 * DM)           // 3072
#define LWIN  128
#define GSTR  64

typedef __attribute__((ext_vector_type(16))) __bf16        v16bf;
typedef __attribute__((ext_vector_type(8)))  float         v8f;
typedef __attribute__((ext_vector_type(4)))  float         v4f;
typedef __attribute__((ext_vector_type(4)))  unsigned int  u32x4;

union Frag {
    v16bf          v;
    u32x4          q[2];
    unsigned short h[16];
};

__device__ __forceinline__ unsigned short f2bf(float f) {
    union { float f; unsigned u; } x; x.f = f;
    unsigned r = x.u + 0x7FFFu + ((x.u >> 16) & 1u);   // round-to-nearest-even
    return (unsigned short)(r >> 16);
}

__device__ __forceinline__ v8f wmma_bf16(v16bf a, v16bf b, v8f c) {
    return __builtin_amdgcn_wmma_f32_16x16x32_bf16(false, a, false, b, (short)0, c, false, false);
}

// ---------- stage 0a: fp32 -> bf16 (row-major copy, for X) ----------
__global__ void cvt_bf16_kernel(const float* __restrict__ in,
                                unsigned short* __restrict__ out, int n) {
    int i = blockIdx.x * blockDim.x + threadIdx.x;
    if (i < n) out[i] = f2bf(in[i]);
}

// ---------- stage 0b: fp32 -> bf16 with transpose (for weights, N-major) ----------
// in[rows][cols] f32  ->  out[cols][rows] bf16
__global__ void cvt_bf16_t_kernel(const float* __restrict__ in,
                                  unsigned short* __restrict__ out,
                                  int rows, int cols) {
    int i = blockIdx.x * blockDim.x + threadIdx.x;
    if (i < rows * cols) {
        int r = i / cols, c = i - r * cols;
        out[(size_t)c * rows + r] = f2bf(in[i]);
    }
}

// ---------- stage 0c: RoPE tables cos/sin [SEQ][32] ----------
__global__ void rope_tab_kernel(float* __restrict__ ct, float* __restrict__ st) {
    int i = blockIdx.x * blockDim.x + threadIdx.x;   // 0 .. SEQ*32-1
    if (i < SEQ * 32) {
        int t = i >> 5, f = i & 31;
        float inv = __powf(10000.0f, -(float)f * (1.0f / 32.0f));
        float sn, cs;
        __sincosf((float)t * inv, &sn, &cs);
        ct[i] = cs; st[i] = sn;
    }
}

// ---------- stage 1: QKV GEMM + RoPE epilogue ----------
// X[8192,1024]bf16 @ WqkvT[3072,1024]bf16(N-major) -> q/k [B,H,T,DH], v [B,H,DH,T]
__global__ __launch_bounds__(128) void qkv_rope_kernel(
        const unsigned short* __restrict__ xb,
        const unsigned short* __restrict__ wT,
        const float* __restrict__ ct, const float* __restrict__ st,
        unsigned short* __restrict__ qb,
        unsigned short* __restrict__ kb,
        unsigned short* __restrict__ vb) {
    const int lane = threadIdx.x & 31;
    const int wave = threadIdx.x >> 5;
    const int l15  = lane & 15;
    const int l2   = lane >> 4;
    const int m0   = blockIdx.x * 128 + wave * 32;   // wave M base (32 rows)
    const int n0   = blockIdx.y * 64;                // block N base (one head section)

    v8f acc[2][4] = {};
    for (int k0 = 0; k0 < DM; k0 += 32) {
        Frag a[2];
#pragma unroll
        for (int mt = 0; mt < 2; ++mt) {
            const unsigned short* ap = xb + (size_t)(m0 + mt * 16 + l15) * DM + k0 + l2 * 8;
            a[mt].q[0] = *(const u32x4*)(ap);
            a[mt].q[1] = *(const u32x4*)(ap + 16);
        }
#pragma unroll
        for (int nt = 0; nt < 4; ++nt) {
            Frag bfr;   // WqkvT row-major [N][K]: lane gets K in table order (8h.., 16+8h..) at fixed N
            const unsigned short* bp = wT + (size_t)(n0 + nt * 16 + l15) * DM + k0 + l2 * 8;
            bfr.q[0] = *(const u32x4*)(bp);
            bfr.q[1] = *(const u32x4*)(bp + 16);
            acc[0][nt] = wmma_bf16(a[0].v, bfr.v, acc[0][nt]);
            acc[1][nt] = wmma_bf16(a[1].v, bfr.v, acc[1][nt]);
        }
    }

    const int sec = n0 >> 10;            // 0=q 1=k 2=v (uniform per block)
    const int hh  = (n0 >> 6) & 15;      // head       (uniform per block)
#pragma unroll
    for (int mt = 0; mt < 2; ++mt) {
#pragma unroll
        for (int nt = 0; nt < 4; ++nt) {
#pragma unroll
            for (int r = 0; r < 8; ++r) {
                int m  = m0 + mt * 16 + l2 * 8 + r;
                int d  = nt * 16 + l15;
                int bb = m >> 11;
                int t  = m & (SEQ - 1);
                float val = acc[mt][nt][r];
                if (sec < 2) {           // RoPE via tables (uniform branch)
                    float other = __shfl_xor(val, 1, 32);
                    float cs = ct[t * 32 + (d >> 1)];
                    float sn = st[t * 32 + (d >> 1)];
                    val = (d & 1) ? (other * sn + val * cs) : (val * cs - other * sn);
                    unsigned short* dst = (sec == 0) ? qb : kb;
                    dst[((size_t)(bb * NH + hh) * SEQ + t) * DH + d] = f2bf(val);
                } else {                 // V stored transposed: [B,H,DH,T]
                    vb[((size_t)(bb * NH + hh) * DH + d) * SEQ + t] = f2bf(val);
                }
            }
        }
    }
}

// ---------- stage 2: flash attention with local+global causal mask ----------
// one wave per (b,h, 16-query tile); online softmax; P re-layout via LDS
__global__ __launch_bounds__(32) void attn_kernel(
        const unsigned short* __restrict__ qb,
        const unsigned short* __restrict__ kb,
        const unsigned short* __restrict__ vb,    // [B,H,DH,T]
        unsigned short* __restrict__ ab) {
    __shared__ float P[16 * 32];

    const int lane = threadIdx.x & 31;
    const int l15  = lane & 15;
    const int l2   = lane >> 4;
    const int bh   = blockIdx.y;
    const int qt0  = blockIdx.x * 16;
    const int bb   = bh >> 4;
    const int hh   = bh & 15;
    const unsigned short* Q = qb + (size_t)bh * SEQ * DH;
    const unsigned short* K = kb + (size_t)bh * SEQ * DH;
    const unsigned short* V = vb + (size_t)bh * DH * SEQ;

    // Q A-fragments for both 32-wide K-chunks of dh=64
    Frag aq[2];
#pragma unroll
    for (int c = 0; c < 2; ++c) {
        const unsigned short* qp = Q + (qt0 + l15) * DH + c * 32 + l2 * 8;
        aq[c].q[0] = *(const u32x4*)(qp);
        aq[c].q[1] = *(const u32x4*)(qp + 16);
    }

    float mi[8], li[8];
    v8f o[4] = {};
#pragma unroll
    for (int r = 0; r < 8; ++r) { mi[r] = -1e30f; li[r] = 0.0f; }

    const bool qtile_glob = (qt0 & (GSTR - 1)) == 0;
    const float scale = 0.125f;

    for (int kt0 = 0; kt0 <= qt0 + 15; kt0 += 32) {
        // tile-level sparsity skip
        if (!(qtile_glob || ((kt0 & (GSTR - 1)) == 0) || (kt0 + 158 >= qt0))) continue;

        // prefetch next tile's K/V (-> global_prefetch_b8)
        if (kt0 + 32 <= qt0 + 15) {
            __builtin_prefetch(K + (kt0 + 32 + l15) * DH, 0, 3);
            __builtin_prefetch(V + l15 * SEQ + kt0 + 32, 0, 3);
        }

        // S = Q @ K^T : two 16x16 f32 tiles over 32 keys
        v8f s0 = {}, s1 = {};
#pragma unroll
        for (int c = 0; c < 2; ++c) {
            Frag bk;
            const unsigned short* kp0 = K + (kt0 + l15) * DH + c * 32 + l2 * 8;
            bk.q[0] = *(const u32x4*)(kp0);
            bk.q[1] = *(const u32x4*)(kp0 + 16);
            s0 = wmma_bf16(aq[c].v, bk.v, s0);
            const unsigned short* kp1 = K + (kt0 + 16 + l15) * DH + c * 32 + l2 * 8;
            bk.q[0] = *(const u32x4*)(kp1);
            bk.q[1] = *(const u32x4*)(kp1 + 16);
            s1 = wmma_bf16(aq[c].v, bk.v, s1);
        }

        // mask + row max
        float mt[8];
#pragma unroll
        for (int r = 0; r < 8; ++r) {
            int qt  = qt0 + l2 * 8 + r;
            int kta = kt0 + l15;
            int ktb = kt0 + 16 + l15;
            bool qg = (qt & (GSTR - 1)) == 0;
            float sa = s0[r] * scale, sb = s1[r] * scale;
            bool ala = (kta <= qt) && ((qt - kta <= LWIN - 1) || ((kta & (GSTR - 1)) == 0) || qg);
            bool alb = (ktb <= qt) && ((qt - ktb <= LWIN - 1) || ((ktb & (GSTR - 1)) == 0) || qg);
            sa = ala ? sa : -1e9f;
            sb = alb ? sb : -1e9f;
            s0[r] = sa; s1[r] = sb;
            float mr = fmaxf(sa, sb);
#pragma unroll
            for (int off = 1; off < 16; off <<= 1)
                mr = fmaxf(mr, __shfl_xor(mr, off, 32));
            mt[r] = mr;
        }

        // online softmax update; stage P(f32) into LDS
        float alpha[8];
#pragma unroll
        for (int r = 0; r < 8; ++r) {
            float nm = fmaxf(mi[r], mt[r]);
            alpha[r] = __expf(mi[r] - nm);
            mi[r] = nm;
            float pa = __expf(s0[r] - nm);
            float pb = __expf(s1[r] - nm);
            float rs = pa + pb;
#pragma unroll
            for (int off = 1; off < 16; off <<= 1)
                rs += __shfl_xor(rs, off, 32);
            li[r] = li[r] * alpha[r] + rs;
            int row = l2 * 8 + r;
            P[row * 32 + l15]      = pa;
            P[row * 32 + 16 + l15] = pb;
        }
        asm volatile("s_wait_dscnt 0" ::: "memory");

        // re-read P in A-layout (vectorized) and convert to bf16
        Frag pf;
        {
            int base = l15 * 32 + l2 * 8;
            v4f c0 = *(const v4f*)&P[base];
            v4f c1 = *(const v4f*)&P[base + 4];
            v4f c2 = *(const v4f*)&P[base + 16];
            v4f c3 = *(const v4f*)&P[base + 20];
#pragma unroll
            for (int j = 0; j < 4; ++j) {
                pf.h[j]      = f2bf(c0[j]);
                pf.h[4 + j]  = f2bf(c1[j]);
                pf.h[8 + j]  = f2bf(c2[j]);
                pf.h[12 + j] = f2bf(c3[j]);
            }
        }
        asm volatile("" ::: "memory");

        // O = O*alpha + P @ V   (V is [DH][T]: contiguous keys per lane)
#pragma unroll
        for (int nt = 0; nt < 4; ++nt) {
#pragma unroll
            for (int r = 0; r < 8; ++r) o[nt][r] *= alpha[r];
        }
#pragma unroll
        for (int nt = 0; nt < 4; ++nt) {
            Frag bv;
            const unsigned short* vp = V + (size_t)(nt * 16 + l15) * SEQ + kt0 + l2 * 8;
            bv.q[0] = *(const u32x4*)(vp);
            bv.q[1] = *(const u32x4*)(vp + 16);
            o[nt] = wmma_bf16(pf.v, bv.v, o[nt]);
        }
    }

    // normalize + write [B,T,C] bf16 for the output projection
#pragma unroll
    for (int nt = 0; nt < 4; ++nt) {
#pragma unroll
        for (int r = 0; r < 8; ++r) {
            int t   = qt0 + l2 * 8 + r;
            int col = hh * DH + nt * 16 + l15;
            ab[((size_t)bb * SEQ + t) * DM + col] = f2bf(o[nt][r] / li[r]);
        }
    }
}

// ---------- stage 3: output projection + bias ----------
__global__ __launch_bounds__(128) void out_proj_kernel(
        const unsigned short* __restrict__ ab,
        const unsigned short* __restrict__ wT,   // WoutT [N][K] bf16
        const float* __restrict__ bout,
        float* __restrict__ out) {
    const int lane = threadIdx.x & 31;
    const int wave = threadIdx.x >> 5;
    const int l15  = lane & 15;
    const int l2   = lane >> 4;
    const int m0   = blockIdx.x * 128 + wave * 32;
    const int n0   = blockIdx.y * 64;

    v8f acc[2][4] = {};
    for (int k0 = 0; k0 < DM; k0 += 32) {
        Frag a[2];
#pragma unroll
        for (int mt = 0; mt < 2; ++mt) {
            const unsigned short* ap = ab + (size_t)(m0 + mt * 16 + l15) * DM + k0 + l2 * 8;
            a[mt].q[0] = *(const u32x4*)(ap);
            a[mt].q[1] = *(const u32x4*)(ap + 16);
        }
#pragma unroll
        for (int nt = 0; nt < 4; ++nt) {
            Frag bfr;
            const unsigned short* bp = wT + (size_t)(n0 + nt * 16 + l15) * DM + k0 + l2 * 8;
            bfr.q[0] = *(const u32x4*)(bp);
            bfr.q[1] = *(const u32x4*)(bp + 16);
            acc[0][nt] = wmma_bf16(a[0].v, bfr.v, acc[0][nt]);
            acc[1][nt] = wmma_bf16(a[1].v, bfr.v, acc[1][nt]);
        }
    }
#pragma unroll
    for (int mt = 0; mt < 2; ++mt) {
#pragma unroll
        for (int nt = 0; nt < 4; ++nt) {
#pragma unroll
            for (int r = 0; r < 8; ++r) {
                int m = m0 + mt * 16 + l2 * 8 + r;
                int n = n0 + nt * 16 + l15;
                out[(size_t)m * DM + n] = acc[mt][nt][r] + bout[n];
            }
        }
    }
}

// ---------- launch ----------
extern "C" void kernel_launch(void* const* d_in, const int* in_sizes, int n_in,
                              void* d_out, int out_size, void* d_ws, size_t ws_size,
                              hipStream_t stream) {
    const float* x    = (const float*)d_in[0];
    const float* Wqkv = (const float*)d_in[1];
    const float* Wout = (const float*)d_in[2];
    const float* bout = (const float*)d_in[3];
    float* out = (float*)d_out;

    // workspace carve-up
    unsigned short* xb    = (unsigned short*)d_ws;
    unsigned short* wqkvT = xb    + (size_t)MTOT * DM;       // [3072][1024]
    unsigned short* woutT = wqkvT + (size_t)DM * N_QKV;      // [1024][1024]
    unsigned short* qb    = woutT + (size_t)DM * DM;
    unsigned short* kb    = qb    + (size_t)MTOT * DM;
    unsigned short* vb    = kb    + (size_t)MTOT * DM;       // transposed [B,H,DH,T]
    unsigned short* abuf  = vb    + (size_t)MTOT * DM;
    float*          ct    = (float*)(abuf + (size_t)MTOT * DM);
    float*          st    = ct + SEQ * 32;

    const int nx = MTOT * DM, nw = DM * N_QKV, no = DM * DM, nt = SEQ * 32;
    cvt_bf16_kernel  <<<(nx + 255) / 256, 256, 0, stream>>>(x, xb, nx);
    cvt_bf16_t_kernel<<<(nw + 255) / 256, 256, 0, stream>>>(Wqkv, wqkvT, DM, N_QKV);
    cvt_bf16_t_kernel<<<(no + 255) / 256, 256, 0, stream>>>(Wout, woutT, DM, DM);
    rope_tab_kernel  <<<(nt + 255) / 256, 256, 0, stream>>>(ct, st);

    qkv_rope_kernel<<<dim3(MTOT / 128, N_QKV / 64), 128, 0, stream>>>(
        xb, wqkvT, ct, st, qb, kb, vb);
    attn_kernel<<<dim3(SEQ / 16, BATCH * NH), 32, 0, stream>>>(qb, kb, vb, abuf);
    out_proj_kernel<<<dim3(MTOT / 128, DM / 64), 128, 0, stream>>>(abuf, woutT, bout, out);
}
